// dpnn6_79027398246429
// MI455X (gfx1250) — matrix operand order repaired and checked by reading, not compiled, not run on hardware
//
#include <hip/hip_runtime.h>

// ---------------------------------------------------------------------------
// Types for CDNA5 WMMA (wave32): v_wmma_f32_16x16x32_bf16
// ---------------------------------------------------------------------------
typedef __bf16 bf16_t;
typedef __attribute__((ext_vector_type(16))) __bf16 v16bf;
typedef __attribute__((ext_vector_type(8)))  __bf16 v8bf;
typedef __attribute__((ext_vector_type(8)))  float  v8f;

static __device__ __forceinline__ bf16_t f2bf(float f) {
    unsigned int u = __builtin_bit_cast(unsigned int, f);
    unsigned int r = u + 0x7FFFu + ((u >> 16) & 1u);     // round-to-nearest-even
    unsigned short h = (unsigned short)(r >> 16);
    return __builtin_bit_cast(bf16_t, h);
}

// ---------------------------------------------------------------------------
// CDNA5 async cache->LDS copy (no VGPR round trip), tracked by ASYNCcnt.
// VDST = 32-bit LDS byte address (low 32 bits of generic pointer to __shared__),
// VADDR = 64-bit global address, saddr = off (GV mode).
// ---------------------------------------------------------------------------
static __device__ __forceinline__ void async_load_b128(unsigned lds_addr, const void* gptr) {
    asm volatile("global_load_async_to_lds_b128 %0, %1, off"
                 :: "v"(lds_addr), "v"(gptr)
                 : "memory");
}
static __device__ __forceinline__ void wait_asynccnt0() {
    asm volatile("s_wait_asynccnt 0" ::: "memory");
}

// ---------------------------------------------------------------------------
// fp32 -> bf16 conversion (grid-stride)
// ---------------------------------------------------------------------------
__global__ __launch_bounds__(256) void f32_to_bf16_kernel(
        const float* __restrict__ src, bf16_t* __restrict__ dst, long long n) {
    long long i = (long long)blockIdx.x * blockDim.x + threadIdx.x;
    long long stride = (long long)gridDim.x * blockDim.x;
    for (; i < n; i += stride) dst[i] = f2bf(src[i]);
}

// ---------------------------------------------------------------------------
// Generic batched NT GEMM:  C[b] = alpha * A[b] (MxK) * B[b]^T (NxK) (+ bias[N])
// A row-major, B row-major, both bf16. Output f32 or bf16, optionally stored
// transposed (C[n*ldc + m]).
// Block tile 128x128, K-step 32, 256 threads = 8 waves (4x2), each wave does
// a 32x64 patch = 2x4 WMMA 16x16 tiles. Staging uses async LDS loads.
// ---------------------------------------------------------------------------
#define BM 128
#define BN 128
#define BK 32
#define LDS_STRIDE 40   // 32 halfwords + 8 pad (80B rows, 16B aligned)

template<bool OUT_BF16, bool USE_BIAS, bool TRANS_STORE>
__global__ __launch_bounds__(256) void gemm_nt_wmma(
        const bf16_t* __restrict__ A, long long lda, long long aBatch,
        const bf16_t* __restrict__ Bm, long long ldb, long long bBatch,
        void* __restrict__ Cv, long long ldc, long long cBatch,
        const float* __restrict__ bias, float alpha, int Kdim)
{
    __shared__ bf16_t As[2][BM * LDS_STRIDE];
    __shared__ bf16_t Bs[2][BN * LDS_STRIDE];

    const int tid  = threadIdx.x;
    const int lane = tid & 31;
    const int wave = tid >> 5;     // 0..7
    const int wr   = wave & 3;     // wave row (M): 4
    const int wc   = wave >> 2;    // wave col (N): 2
    const int sub  = lane & 15;
    const int grp  = lane >> 4;

    const long long bm = (long long)blockIdx.x * BM;
    const long long bn = (long long)blockIdx.y * BN;
    const int b = blockIdx.z;
    A  += (long long)b * aBatch;
    Bm += (long long)b * bBatch;

    const int kTiles = Kdim / BK;

    v8f acc[2][4];
#pragma unroll
    for (int mt = 0; mt < 2; ++mt)
#pragma unroll
        for (int nt = 0; nt < 4; ++nt)
            acc[mt][nt] = {};

    // Issue cache->LDS async copies for one 128x32 A tile and 128x32 B tile.
    // 512 16B chunks per operand; 2 A-chunks + 2 B-chunks per thread.
    auto issue_tiles = [&](int buf, int kt) {
        const long long k0 = (long long)kt * BK;
#pragma unroll
        for (int it = 0; it < 2; ++it) {
            int c   = it * 256 + tid;
            int row = c >> 2;
            int col = c & 3;
            async_load_b128(
                (unsigned)(uintptr_t)&As[buf][row * LDS_STRIDE + col * 8],
                &A[(bm + row) * lda + k0 + col * 8]);
            async_load_b128(
                (unsigned)(uintptr_t)&Bs[buf][row * LDS_STRIDE + col * 8],
                &Bm[(bn + row) * ldb + k0 + col * 8]);
        }
        // Speculative prefetch of the tile after next (global_prefetch_b8)
        if (kt + 2 < kTiles) {
            const long long kp = (long long)(kt + 2) * BK;
            __builtin_prefetch(&A[(bm + (tid >> 1)) * lda + kp], 0, 0);
            __builtin_prefetch(&Bm[(bn + (tid >> 1)) * ldb + kp], 0, 0);
        }
    };

    auto compute = [&](int buf) {
        v16bf afrag[2];
        v16bf bfrag[4];
        // A fragment: lane-group 0 -> K {0..7, 16..23}; group 1 -> K {8..15, 24..31}
#pragma unroll
        for (int mt = 0; mt < 2; ++mt) {
            const bf16_t* p = &As[buf][(wr * 32 + mt * 16 + sub) * LDS_STRIDE + grp * 8];
            v8bf lo = *(const v8bf*)p;
            v8bf hi = *(const v8bf*)(p + 16);
            afrag[mt] = __builtin_shufflevector(lo, hi,
                0, 1, 2, 3, 4, 5, 6, 7, 8, 9, 10, 11, 12, 13, 14, 15);
        }
        // B fragment: lane-group 0 -> K {0..7, 16..23}; group 1 -> K {8..15, 24..31}
#pragma unroll
        for (int nt = 0; nt < 4; ++nt) {
            const bf16_t* p = &Bs[buf][(wc * 64 + nt * 16 + sub) * LDS_STRIDE + grp * 8];
            v8bf lo = *(const v8bf*)p;
            v8bf hi = *(const v8bf*)(p + 16);
            bfrag[nt] = __builtin_shufflevector(lo, hi,
                0, 1, 2, 3, 4, 5, 6, 7, 8, 9, 10, 11, 12, 13, 14, 15);
        }
#pragma unroll
        for (int mt = 0; mt < 2; ++mt)
#pragma unroll
            for (int nt = 0; nt < 4; ++nt)
                acc[mt][nt] = __builtin_amdgcn_wmma_f32_16x16x32_bf16(
                    false, afrag[mt], false, bfrag[nt],
                    (short)0, acc[mt][nt], false, false);
    };

    issue_tiles(0, 0);
    wait_asynccnt0();
    __syncthreads();
    for (int kt = 0; kt < kTiles; ++kt) {
        const int cur = kt & 1;
        if (kt + 1 < kTiles) issue_tiles(cur ^ 1, kt + 1);   // overlap with WMMAs
        compute(cur);
        wait_asynccnt0();        // next buffer landed in LDS
        __syncthreads();
    }

    // Epilogue. C/D layout: VGPR r <-> row 8*grp + r, col = lane%16.
    float*  Cf = (float*)Cv;
    bf16_t* Cb = (bf16_t*)Cv;
    const long long cbase = (long long)b * cBatch;
#pragma unroll
    for (int mt = 0; mt < 2; ++mt) {
#pragma unroll
        for (int nt = 0; nt < 4; ++nt) {
            const long long tr = bm + wr * 32 + mt * 16 + grp * 8;
            const long long tc = bn + wc * 64 + nt * 16 + sub;
            const float bval = USE_BIAS ? bias[tc] : 0.0f;
#pragma unroll
            for (int r = 0; r < 8; ++r) {
                float v = acc[mt][nt][r] * alpha + bval;
                const long long rr = tr + r;
                const long long idx = TRANS_STORE ? (tc * ldc + rr) : (rr * ldc + tc);
                if (OUT_BF16) Cb[cbase + idx] = f2bf(v);
                else          Cf[cbase + idx] = v;
            }
        }
    }
}

// ---------------------------------------------------------------------------
// Row softmax over dist[b,i,:] (2048 wide), written TRANSPOSED as bf16:
// wT[b,k,i] = softmax_k(dist[b,i,k]). One 256-thread block per (b,i) row.
// ---------------------------------------------------------------------------
#define ROWLEN 2048
#define NKROWS 2048

__global__ __launch_bounds__(256) void softmax_rows_transpose(
        const float* __restrict__ dist, bf16_t* __restrict__ wT)
{
    __shared__ float red[256];
    const int tid = threadIdx.x;
    const long long row = blockIdx.x;           // b*NK + i
    const long long b = row / NKROWS;
    const long long i = row - b * NKROWS;
    const float* src = dist + row * (long long)ROWLEN;

    float vals[8];
    float m = -3.4e38f;
#pragma unroll
    for (int j = 0; j < 8; ++j) {
        vals[j] = src[j * 256 + tid];
        m = fmaxf(m, vals[j]);
    }
    red[tid] = m; __syncthreads();
    for (int s = 128; s > 0; s >>= 1) {
        if (tid < s) red[tid] = fmaxf(red[tid], red[tid + s]);
        __syncthreads();
    }
    m = red[0]; __syncthreads();

    float sum = 0.0f;
#pragma unroll
    for (int j = 0; j < 8; ++j) {
        vals[j] = __expf(vals[j] - m);
        sum += vals[j];
    }
    red[tid] = sum; __syncthreads();
    for (int s = 128; s > 0; s >>= 1) {
        if (tid < s) red[tid] += red[tid + s];
        __syncthreads();
    }
    const float inv = 1.0f / red[0];

    bf16_t* dstT = wT + b * (long long)ROWLEN * NKROWS + i;   // wT[b][k][i]
#pragma unroll
    for (int j = 0; j < 8; ++j) {
        const long long k = j * 256 + tid;
        dstT[k * NKROWS] = f2bf(vals[j] * inv);
    }
}

// ---------------------------------------------------------------------------
// Orchestration
// ---------------------------------------------------------------------------
extern "C" void kernel_launch(void* const* d_in, const int* in_sizes, int n_in,
                              void* d_out, int out_size, void* d_ws, size_t ws_size,
                              hipStream_t stream) {
    (void)in_sizes; (void)n_in; (void)out_size; (void)ws_size;

    const float* KEY   = (const float*)d_in[0];
    const float* VALUE = (const float*)d_in[1];
    const float* QUERY = (const float*)d_in[2];
    const float* Wk_w  = (const float*)d_in[3];
    const float* Wk_b  = (const float*)d_in[4];
    const float* Wq_w  = (const float*)d_in[5];
    const float* Wq_b  = (const float*)d_in[6];
    const float* Wv_w  = (const float*)d_in[7];
    const float* Wv_b  = (const float*)d_in[8];
    const float* lin_w = (const float*)d_in[9];
    const float* lin_b = (const float*)d_in[10];
    float* out = (float*)d_out;

    constexpr long long Bb = 8, NK = 2048, NQ = 2048, D = 256, H = 1024;

    // Workspace layout (256B-aligned slices)
    char* p = (char*)d_ws;
    auto alloc = [&](long long bytes) -> void* {
        void* r = (void*)p;
        p += (bytes + 255) & ~255LL;
        return r;
    };
    bf16_t* keyb   = (bf16_t*)alloc(Bb * NK * D * 2);
    bf16_t* valb   = (bf16_t*)alloc(Bb * NK * D * 2);
    bf16_t* queryb = (bf16_t*)alloc(Bb * NQ * D * 2);
    bf16_t* wkb    = (bf16_t*)alloc(H * D * 2);
    bf16_t* wqb    = (bf16_t*)alloc(H * D * 2);
    bf16_t* wvb    = (bf16_t*)alloc(H * D * 2);
    bf16_t* linwb  = (bf16_t*)alloc(D * H * 2);
    bf16_t* Pk     = (bf16_t*)alloc(Bb * NK * H * 2);   // [b, i, h]
    bf16_t* Pq     = (bf16_t*)alloc(Bb * NQ * H * 2);   // [b, k, h]
    bf16_t* PvT    = (bf16_t*)alloc(Bb * H * NK * 2);   // [b, h, i]  (transposed V proj)
    float*  dist   = (float*)alloc(Bb * NK * NQ * 4);   // [b, i, k]
    bf16_t* wT     = (bf16_t*)alloc(Bb * NQ * NK * 2);  // [b, k, i]
    bf16_t* ctx    = (bf16_t*)alloc(Bb * NQ * H * 2);   // [b, k, h]

    // 1) fp32 -> bf16 conversions
    f32_to_bf16_kernel<<<4096, 256, 0, stream>>>(KEY,   keyb,   Bb * NK * D);
    f32_to_bf16_kernel<<<4096, 256, 0, stream>>>(VALUE, valb,   Bb * NK * D);
    f32_to_bf16_kernel<<<4096, 256, 0, stream>>>(QUERY, queryb, Bb * NQ * D);
    f32_to_bf16_kernel<<<512,  256, 0, stream>>>(Wk_w,  wkb,    H * D);
    f32_to_bf16_kernel<<<512,  256, 0, stream>>>(Wq_w,  wqb,    H * D);
    f32_to_bf16_kernel<<<512,  256, 0, stream>>>(Wv_w,  wvb,    H * D);
    f32_to_bf16_kernel<<<512,  256, 0, stream>>>(lin_w, linwb,  D * H);

    // 2) Projections: P = X @ W^T + b  (M=2048, N=1024, K=256)
    dim3 gProj(16, 8, 8);
    gemm_nt_wmma<true, true, false><<<gProj, 256, 0, stream>>>(
        keyb, D, NK * D, wkb, D, 0, (void*)Pk, H, NK * H, Wk_b, 1.0f, (int)D);
    gemm_nt_wmma<true, true, false><<<gProj, 256, 0, stream>>>(
        queryb, D, NQ * D, wqb, D, 0, (void*)Pq, H, NQ * H, Wq_b, 1.0f, (int)D);
    // V projection stored transposed: PvT[b, h, i]
    gemm_nt_wmma<true, true, true><<<gProj, 256, 0, stream>>>(
        valb, D, NK * D, wvb, D, 0, (void*)PvT, NK, H * NK, Wv_b, 1.0f, (int)D);

    // 3) dist[b,i,k] = scale * Pk[b,i,:] . Pq[b,k,:]  (M=N=2048, K=1024)
    dim3 gDist(16, 16, 8);
    gemm_nt_wmma<false, false, false><<<gDist, 256, 0, stream>>>(
        Pk, H, NK * H, Pq, H, NQ * H, (void*)dist, NQ, NK * NQ,
        nullptr, 0.03125f /* 1/sqrt(1024) */, (int)H);

    // 4) softmax over k, store transposed bf16 wT[b,k,i]
    softmax_rows_transpose<<<(unsigned)(Bb * NK), 256, 0, stream>>>(dist, wT);

    // 5) context[b,k,h] = sum_i wT[b,k,i] * PvT[b,h,i]  (M=2048, N=1024, K=2048)
    dim3 gCtx(16, 8, 8);
    gemm_nt_wmma<true, false, false><<<gCtx, 256, 0, stream>>>(
        wT, NK, NQ * NK, PvT, NK, H * NK, (void*)ctx, H, NQ * H,
        nullptr, 1.0f, (int)NK);

    // 6) res[b,k,o] = ctx[b,k,:] . lin_w[o,:] + lin_b[o]  (M=2048, N=256, K=1024)
    dim3 gOut(16, 2, 8);
    gemm_nt_wmma<false, true, false><<<gOut, 256, 0, stream>>>(
        ctx, H, NQ * H, linwb, H, 0, (void*)out, D, NQ * D,
        lin_b, 1.0f, (int)H);
}
